// MultiHeadAttention_5480378270208
// MI455X (gfx1250) — matrix operand order repaired and checked by reading, not compiled, not run on hardware
//
#include <hip/hip_runtime.h>
#include <hip/hip_bf16.h>

typedef _Float16 v8h  __attribute__((ext_vector_type(8)));
typedef _Float16 v16h __attribute__((ext_vector_type(16)));
typedef float    v8f  __attribute__((ext_vector_type(8)));

#define WMMA_F16(a, b, c) \
  __builtin_amdgcn_wmma_f32_16x16x32_f16(false, (a), false, (b), (short)0, (c), false, false)

// Scheduling hints: 0x100 = DS-read group, 0x008 = MFMA/WMMA group.
#if defined(__has_builtin)
# if __has_builtin(__builtin_amdgcn_sched_group_barrier)
#  define SCHED_GROUP(m, n, s) __builtin_amdgcn_sched_group_barrier((m), (n), (s))
# endif
#endif
#ifndef SCHED_GROUP
# define SCHED_GROUP(m, n, s)
#endif

static __device__ __forceinline__ v16h make_v16(v8h lo, v8h hi) {
  return __builtin_shufflevector(lo, hi, 0,1,2,3,4,5,6,7,8,9,10,11,12,13,14,15);
}

constexpr int Bn   = 64;
constexpr int T    = 256;
constexpr int Cc   = 384;
constexpr int H    = 6;
constexpr int Dh   = 64;
constexpr int Mtot = Bn * T;          // 16384
constexpr int HCD  = H * Cc * Dh;     // 147456
constexpr int KT   = Cc / 32;         // 12 k-steps

// ---------------------------------------------------------------- converts
__global__ void f32_to_f16(const float* __restrict__ src,
                           _Float16* __restrict__ dst, int n) {
  int i = blockIdx.x * blockDim.x + threadIdx.x;
  if (i < n) dst[i] = (_Float16)src[i];
}

// ---------------------------------------------------------------- QKV GEMM
// out[z][b][h][t][d] = sum_c x[b,t,c] * W[z,h,c,d]
// 256 threads, 8 waves; each wave owns TWO 16-row tiles (rows w*16 and 128+w*16).
__global__ __launch_bounds__(256) void qkv_gemm(
    const _Float16* __restrict__ xh,    // [Mtot, C]
    const _Float16* __restrict__ wqkv,  // [3, H, C, D]
    _Float16* __restrict__ qkv)         // [3, B, H, T, D]
{
  constexpr int LSTR = 40;  // f16 stride, 80B: 16B aligned + conflict pad
  __shared__ __align__(16) _Float16 xs[256 * LSTR];
  __shared__ __align__(16) _Float16 wt[64 * LSTR];   // W tile, transposed

  const int tid  = threadIdx.x;
  const int lane = tid & 31;
  const int w    = tid >> 5;      // wave 0..7
  const int half = lane >> 4;
  const int ln   = lane & 15;
  const int M0   = blockIdx.x * 256;
  const int h    = blockIdx.y;
  const int z    = blockIdx.z;
  const _Float16* W = wqkv + (size_t)(z * H + h) * Cc * Dh;

  const int srow = tid >> 1, sh2 = tid & 1;   // x-tile staging role
  const int wd = tid & 63, cg = tid >> 6;     // W-tile staging role

  v8f acc[2][4] = {};

  // register double-buffer: prefetch next global tile during compute
  v8h xr[2][2];
  _Float16 wr[8];
  auto load_glb = [&](int kk) {
    #pragma unroll
    for (int mt = 0; mt < 2; ++mt) {
      const _Float16* xsrc =
          xh + (size_t)(M0 + mt * 128 + srow) * Cc + kk + sh2 * 16;
      xr[mt][0] = *(const v8h*)(xsrc);
      xr[mt][1] = *(const v8h*)(xsrc + 8);
    }
    #pragma unroll
    for (int j = 0; j < 8; ++j)
      wr[j] = W[(size_t)(kk + cg * 8 + j) * Dh + wd];
  };

  load_glb(0);

  for (int kt = 0; kt < KT; ++kt) {
    __syncthreads();
    #pragma unroll
    for (int mt = 0; mt < 2; ++mt) {
      *(v8h*)&xs[(mt * 128 + srow) * LSTR + sh2 * 16]     = xr[mt][0];
      *(v8h*)&xs[(mt * 128 + srow) * LSTR + sh2 * 16 + 8] = xr[mt][1];
    }
    #pragma unroll
    for (int j = 0; j < 8; ++j) wt[wd * LSTR + cg * 8 + j] = wr[j];
    __syncthreads();

    if (kt + 1 < KT) load_glb((kt + 1) * 32);   // overlaps with WMMAs below

    v16h A[2];
    #pragma unroll
    for (int mt = 0; mt < 2; ++mt) {
      int arow = mt * 128 + w * 16 + ln;
      A[mt] = make_v16(*(v8h*)&xs[arow * LSTR + half * 8],
                       *(v8h*)&xs[arow * LSTR + 16 + half * 8]);
    }
    v16h Bf[4];
    #pragma unroll
    for (int n = 0; n < 4; ++n) {
      int d = n * 16 + ln;
      Bf[n] = make_v16(*(v8h*)&wt[d * LSTR + half * 8],
                       *(v8h*)&wt[d * LSTR + 16 + half * 8]);
    }
    #pragma unroll
    for (int n = 0; n < 4; ++n)
      #pragma unroll
      for (int mt = 0; mt < 2; ++mt)
        acc[mt][n] = WMMA_F16(A[mt], Bf[n], acc[mt][n]);

    // all 12 DS reads first, then the 8-WMMA chain
    SCHED_GROUP(0x100, 12, 0);
    SCHED_GROUP(0x008, 8, 0);
  }

  #pragma unroll
  for (int mt = 0; mt < 2; ++mt)
    #pragma unroll
    for (int n = 0; n < 4; ++n)
      #pragma unroll
      for (int r = 0; r < 8; ++r) {
        int m = M0 + mt * 128 + w * 16 + r + 8 * half;
        int b = m >> 8, t = m & 255;
        int d = n * 16 + ln;
        qkv[(size_t)(((z * Bn + b) * H + h) * T + t) * Dh + d] =
            (_Float16)acc[mt][n][r];
      }
}

// ---------------------------------------------------------------- attention
__global__ __launch_bounds__(128) void attn(
    const _Float16* __restrict__ qkv,   // [3, B, H, T, D]
    _Float16* __restrict__ aout)        // [B, T, H*D]
{
  __shared__ __align__(16) _Float16 ks[T * Dh];   // K, later reused for wei
  __shared__ __align__(16) _Float16 vt[Dh * T];   // V transposed: vt[d][s]

  const int tid  = threadIdx.x;
  const int lane = tid & 31;
  const int w    = tid >> 5;     // wave 0..3 -> 16-row tile
  const int half = lane >> 4;
  const int ln   = lane & 15;
  const int b = blockIdx.z, h = blockIdx.y;
  const int r0 = blockIdx.x * 64;

  const _Float16* Qg = qkv + (size_t)(((0 * Bn + b) * H + h) * T) * Dh;
  const _Float16* Kg = qkv + (size_t)(((1 * Bn + b) * H + h) * T) * Dh;
  const _Float16* Vg = qkv + (size_t)(((2 * Bn + b) * H + h) * T) * Dh;

  // stage K (vector copy, layout-preserving)
  #pragma unroll
  for (int j = 0; j < 16; ++j) {
    int o = tid * 128 + j * 8;
    *(v8h*)&ks[o] = *(const v8h*)&Kg[o];
  }
  // stage V transposed: each thread handles 2 rows, vectorized global reads,
  // bank-conflict-free scalar LDS scatter (bank == tid % 64).
  #pragma unroll
  for (int rr = 0; rr < 2; ++rr) {
    int s = tid * 2 + rr;
    #pragma unroll
    for (int j = 0; j < 8; ++j) {
      v8h vv = *(const v8h*)&Vg[(size_t)s * Dh + j * 8];
      #pragma unroll
      for (int e = 0; e < 8; ++e) vt[(j * 8 + e) * T + s] = vv[e];
    }
  }
  __syncthreads();

  // phase 1: scores = q @ k^T  (16 x 256 strip per wave)
  v8f acc[16] = {};
  #pragma unroll
  for (int dk = 0; dk < 2; ++dk) {
    int d0 = dk * 32;
    int t = r0 + w * 16 + ln;
    const _Float16* qrow = Qg + (size_t)t * Dh + d0;
    v16h A = make_v16(*(const v8h*)(qrow + half * 8),
                      *(const v8h*)(qrow + 16 + half * 8));
    #pragma unroll
    for (int cb = 0; cb < 4; ++cb) {          // groups of 4 col-tiles
      v16h Bf[4];
      #pragma unroll
      for (int u = 0; u < 4; ++u) {
        int s = (cb * 4 + u) * 16 + ln;
        Bf[u] = make_v16(*(v8h*)&ks[s * Dh + d0 + half * 8],
                         *(v8h*)&ks[s * Dh + d0 + 16 + half * 8]);
      }
      #pragma unroll
      for (int u = 0; u < 4; ++u)
        acc[cb * 4 + u] = WMMA_F16(A, Bf[u], acc[cb * 4 + u]);
      SCHED_GROUP(0x100, 8, 0);
      SCHED_GROUP(0x008, 4, 0);
    }
  }

  // causal softmax, fully in registers (row lives in one 16-lane half)
  const float scale = 0.125f;  // 1/sqrt(64)
  #pragma unroll
  for (int r = 0; r < 8; ++r) {
    int trow = r0 + w * 16 + r + 8 * half;
    float v[16];
    float mx = -1e30f;
    #pragma unroll
    for (int ct = 0; ct < 16; ++ct) {
      int sg = ct * 16 + ln;
      v[ct] = (sg <= trow) ? acc[ct][r] * scale : -1e30f;
      mx = fmaxf(mx, v[ct]);
    }
    #pragma unroll
    for (int off = 1; off < 16; off <<= 1)
      mx = fmaxf(mx, __shfl_xor(mx, off, 16));
    float sum = 0.f;
    #pragma unroll
    for (int ct = 0; ct < 16; ++ct) { v[ct] = __expf(v[ct] - mx); sum += v[ct]; }
    #pragma unroll
    for (int off = 1; off < 16; off <<= 1)
      sum += __shfl_xor(sum, off, 16);
    float inv = 1.f / sum;
    #pragma unroll
    for (int ct = 0; ct < 16; ++ct) acc[ct][r] = v[ct] * inv;
  }

  __syncthreads();   // all waves done reading ks as K
  // write wei (f16, A-layout source) into ks region: wei[rowl][s]
  #pragma unroll
  for (int r = 0; r < 8; ++r) {
    int rowl = w * 16 + r + 8 * half;
    #pragma unroll
    for (int ct = 0; ct < 16; ++ct)
      ks[rowl * T + ct * 16 + ln] = (_Float16)acc[ct][r];
  }
  __syncthreads();

  // phase 2: out = wei @ V
  v8f oacc[4] = {};
  const int rowA = w * 16 + ln;
  #pragma unroll
  for (int ss = 0; ss < 8; ++ss) {
    int s0 = ss * 32;
    v16h A = make_v16(*(v8h*)&ks[rowA * T + s0 + half * 8],
                      *(v8h*)&ks[rowA * T + s0 + 16 + half * 8]);
    v16h Bf[4];
    #pragma unroll
    for (int n = 0; n < 4; ++n) {
      int d = n * 16 + ln;
      Bf[n] = make_v16(*(v8h*)&vt[d * T + s0 + half * 8],
                       *(v8h*)&vt[d * T + s0 + 16 + half * 8]);
    }
    #pragma unroll
    for (int n = 0; n < 4; ++n) oacc[n] = WMMA_F16(A, Bf[n], oacc[n]);
    SCHED_GROUP(0x100, 10, 0);
    SCHED_GROUP(0x008, 4, 0);
  }

  #pragma unroll
  for (int n = 0; n < 4; ++n) {
    #pragma unroll
    for (int r = 0; r < 8; ++r) {
      int t = r0 + w * 16 + r + 8 * half;
      int col = h * Dh + n * 16 + ln;
      aout[(size_t)(b * T + t) * Cc + col] = (_Float16)oacc[n][r];
    }
  }
}

// ---------------------------------------------------------------- out proj
// y[m, n] = sum_c a[m, c] * Wp[n, c] + bp[n]
// 256 threads, 8 waves, two 16-row M-tiles per wave (256-row block).
__global__ __launch_bounds__(256) void proj_gemm(
    const _Float16* __restrict__ ah,    // [Mtot, C]
    const _Float16* __restrict__ wph,   // [C, C] row-major Wp
    const float* __restrict__ bp,       // [C]
    float* __restrict__ out)            // [Mtot, C]
{
  constexpr int LSTR = 40;
  __shared__ __align__(16) _Float16 as_[256 * LSTR];
  __shared__ __align__(16) _Float16 wps[64 * LSTR];

  const int tid  = threadIdx.x;
  const int lane = tid & 31;
  const int w    = tid >> 5;
  const int half = lane >> 4;
  const int ln   = lane & 15;
  const int M0   = blockIdx.x * 256;
  const int N0   = blockIdx.y * 64;

  const int srow = tid >> 1, sh2 = tid & 1;
  const int nl = tid >> 2, cq = tid & 3;

  v8f acc[2][4] = {};

  v8h ar[2][2], wpr;
  auto load_glb = [&](int kk) {
    #pragma unroll
    for (int mt = 0; mt < 2; ++mt) {
      const _Float16* asrc =
          ah + (size_t)(M0 + mt * 128 + srow) * Cc + kk + sh2 * 16;
      ar[mt][0] = *(const v8h*)(asrc);
      ar[mt][1] = *(const v8h*)(asrc + 8);
    }
    wpr = *(const v8h*)&wph[(size_t)(N0 + nl) * Cc + kk + cq * 8];
  };

  load_glb(0);

  for (int kt = 0; kt < KT; ++kt) {
    __syncthreads();
    #pragma unroll
    for (int mt = 0; mt < 2; ++mt) {
      *(v8h*)&as_[(mt * 128 + srow) * LSTR + sh2 * 16]     = ar[mt][0];
      *(v8h*)&as_[(mt * 128 + srow) * LSTR + sh2 * 16 + 8] = ar[mt][1];
    }
    *(v8h*)&wps[nl * LSTR + cq * 8] = wpr;
    __syncthreads();

    if (kt + 1 < KT) load_glb((kt + 1) * 32);

    v16h A[2];
    #pragma unroll
    for (int mt = 0; mt < 2; ++mt) {
      int arow = mt * 128 + w * 16 + ln;
      A[mt] = make_v16(*(v8h*)&as_[arow * LSTR + half * 8],
                       *(v8h*)&as_[arow * LSTR + 16 + half * 8]);
    }
    v16h Bf[4];
    #pragma unroll
    for (int n = 0; n < 4; ++n) {
      int nn = n * 16 + ln;
      Bf[n] = make_v16(*(v8h*)&wps[nn * LSTR + half * 8],
                       *(v8h*)&wps[nn * LSTR + 16 + half * 8]);
    }
    #pragma unroll
    for (int n = 0; n < 4; ++n)
      #pragma unroll
      for (int mt = 0; mt < 2; ++mt)
        acc[mt][n] = WMMA_F16(A[mt], Bf[n], acc[mt][n]);

    SCHED_GROUP(0x100, 12, 0);
    SCHED_GROUP(0x008, 8, 0);
  }

  #pragma unroll
  for (int mt = 0; mt < 2; ++mt)
    #pragma unroll
    for (int n = 0; n < 4; ++n)
      #pragma unroll
      for (int r = 0; r < 8; ++r) {
        int m = M0 + mt * 128 + w * 16 + r + 8 * half;
        int col = N0 + n * 16 + ln;
        out[(size_t)m * Cc + col] = acc[mt][n][r] + bp[col];
      }
}

// ---------------------------------------------------------------- launcher
extern "C" void kernel_launch(void* const* d_in, const int* in_sizes, int n_in,
                              void* d_out, int out_size, void* d_ws, size_t ws_size,
                              hipStream_t stream) {
  const float* x  = (const float*)d_in[0];
  const float* Wq = (const float*)d_in[1];
  const float* Wk = (const float*)d_in[2];
  const float* Wv = (const float*)d_in[3];
  const float* Wp = (const float*)d_in[4];
  const float* bp = (const float*)d_in[5];
  float* out = (float*)d_out;

  _Float16* wsh = (_Float16*)d_ws;
  size_t o = 0;
  _Float16* xh   = wsh + o; o += (size_t)Mtot * Cc;           // 12.6 MB
  _Float16* wqkv = wsh + o; o += 3 * (size_t)HCD;             // 0.9 MB
  _Float16* wph  = wsh + o; o += (size_t)Cc * Cc;             // 0.3 MB
  _Float16* qkvb = wsh + o; o += 3 * (size_t)Bn * H * T * Dh; // 37.7 MB
  _Float16* aout = wsh + o;                                   // 12.6 MB

  const int n1 = Mtot * Cc;
  f32_to_f16<<<(n1 + 255) / 256, 256, 0, stream>>>(x, xh, n1);
  f32_to_f16<<<(HCD + 255) / 256, 256, 0, stream>>>(Wq, wqkv, HCD);
  f32_to_f16<<<(HCD + 255) / 256, 256, 0, stream>>>(Wk, wqkv + HCD, HCD);
  f32_to_f16<<<(HCD + 255) / 256, 256, 0, stream>>>(Wv, wqkv + 2 * HCD, HCD);
  f32_to_f16<<<(Cc * Cc + 255) / 256, 256, 0, stream>>>(Wp, wph, Cc * Cc);

  qkv_gemm<<<dim3(Mtot / 256, H, 3), 256, 0, stream>>>(xh, wqkv, qkvb);
  attn<<<dim3(T / 64, H, Bn), 128, 0, stream>>>(qkvb, aout);
  proj_gemm<<<dim3(Mtot / 256, Cc / 64), 256, 0, stream>>>(aout, wph, bp, out);
}
